// VisionMamba_Seg_83751862272332
// MI455X (gfx1250) — matrix operand order repaired and checked by reading, not compiled, not run on hardware
//
#include <hip/hip_runtime.h>
#include <hip/hip_bf16.h>
#include <math.h>

// ---------------- model dims ----------------
#define DMODEL 576
#define DINNER 1152
#define DSTATE 16
#define DTRANK 36
#define DCONV  4
#define DEPTH  32
#define LSEQ   784          // NT*HP*HP = 4*14*14
#define XDIM   68           // DT_RANK + 2*D_STATE (logical)
#define XLD    80           // padded leading dim for xdbl (16-float aligned)
#define KPAD   64           // padded K for the Wdt GEMM
#define XZDIM  2304         // 2*D_INNER
#define EPSF   1e-5f

typedef __attribute__((ext_vector_type(16))) __bf16 v16bf;
typedef __attribute__((ext_vector_type(8)))  float  v8f;
typedef __attribute__((ext_vector_type(4)))  float  f4;

__device__ __forceinline__ float sigmoid_f(float x) { return 1.0f / (1.0f + __expf(-x)); }
__device__ __forceinline__ float silu_f(float x)    { return x * sigmoid_f(x); }
__device__ __forceinline__ float softplus_f(float x){ return (x > 20.0f) ? x : log1pf(__expf(x)); }

// ---------------------------------------------------------------------------
// WMMA GEMM:  C[M,N] = A[M,K] * B[N,K]^T    (A,B fp32, bf16-converted in-reg,
// f32 accumulate via v_wmma_f32_16x16x32_bf16).
// REQUIREMENTS: K % 32 == 0; lda, ldb multiples of 4 floats; rows of A/B
// readable for the full padded K (guaranteed by buffer layout).
// Hot loop is 100% branchless: row indices clamped with min(), only the
// epilogue stores are predicated. Each wave computes TWO 16x16 N-tiles
// (reusing its A fragment); 4 waves per block cover 4 M-tiles.
// mode 0: plain store. mode 1: softplus(x + bias[n]).
// ---------------------------------------------------------------------------
__global__ __launch_bounds__(128)
void gemm_wmma_bf16(const float* __restrict__ A,
                    const float* __restrict__ Bw,
                    float* __restrict__ C,
                    const float* __restrict__ bias,
                    int M, int N, int K,
                    int lda, int ldb, int ldc, int mode)
{
    const int wave    = threadIdx.x >> 5;            // 0..3 -> M tile in block
    const int lane    = threadIdx.x & 31;
    const int rowBase = (blockIdx.y * 4 + wave) * 16;
    const int colBase = blockIdx.x * 32;             // two 16-wide N tiles
    const int mRow    = lane & 15;
    const int half    = lane >> 4;                   // 0 or 1
    const int aKh     = half * 16;                   // A K-half offset
    const int bKb     = half * 16;                   // B K-half offset
    const int nCol    = lane & 15;

    const int aRow  = min(rowBase + mRow, M - 1);
    const int bRow0 = min(colBase + nCol, N - 1);
    const int bRow1 = min(colBase + 16 + nCol, N - 1);
    const float* __restrict__ arow = A  + (size_t)aRow  * lda;
    const float* __restrict__ b0   = Bw + (size_t)bRow0 * ldb;
    const float* __restrict__ b1   = Bw + (size_t)bRow1 * ldb;

    v8f acc0 = {}, acc1 = {};
    for (int k0 = 0; k0 < K; k0 += 32) {
        if (k0 + 32 < K) {   // uniform scalar branch; streams weight slabs
            __builtin_prefetch(b0 + k0 + bKb + 32, 0, 1);
            __builtin_prefetch(b1 + k0 + bKb + 32, 0, 1);
        }
        // A fragment: 16 bf16 / lane  (K = aKh..aKh+15)
        f4 a0 = *(const f4*)(arow + k0 + aKh);
        f4 a1 = *(const f4*)(arow + k0 + aKh + 4);
        f4 a2 = *(const f4*)(arow + k0 + aKh + 8);
        f4 a3 = *(const f4*)(arow + k0 + aKh + 12);
        // B fragments: 16 bf16 / lane (K = bKb..bKb+15)
        f4 p0 = *(const f4*)(b0 + k0 + bKb);
        f4 p1 = *(const f4*)(b0 + k0 + bKb + 4);
        f4 p2 = *(const f4*)(b0 + k0 + bKb + 8);
        f4 p3 = *(const f4*)(b0 + k0 + bKb + 12);
        f4 q0 = *(const f4*)(b1 + k0 + bKb);
        f4 q1 = *(const f4*)(b1 + k0 + bKb + 4);
        f4 q2 = *(const f4*)(b1 + k0 + bKb + 8);
        f4 q3 = *(const f4*)(b1 + k0 + bKb + 12);

        v16bf af, bf0, bf1;
#pragma unroll
        for (int j = 0; j < 4; ++j) {
            af[j]      = (__bf16)a0[j];
            af[j + 4]  = (__bf16)a1[j];
            af[j + 8]  = (__bf16)a2[j];
            af[j + 12] = (__bf16)a3[j];
            bf0[j]      = (__bf16)p0[j];
            bf0[j + 4]  = (__bf16)p1[j];
            bf0[j + 8]  = (__bf16)p2[j];
            bf0[j + 12] = (__bf16)p3[j];
            bf1[j]      = (__bf16)q0[j];
            bf1[j + 4]  = (__bf16)q1[j];
            bf1[j + 8]  = (__bf16)q2[j];
            bf1[j + 12] = (__bf16)q3[j];
        }
        acc0 = __builtin_amdgcn_wmma_f32_16x16x32_bf16(
                   false, af, false, bf0, (short)0, acc0, false, false);
        acc1 = __builtin_amdgcn_wmma_f32_16x16x32_bf16(
                   false, af, false, bf1, (short)0, acc1, false, false);
    }

    // ---- epilogue (predicated stores only) ----
    const int col0 = colBase + nCol;
    const int col1 = colBase + 16 + nCol;
    const float bv0 = (mode == 1) ? bias[min(col0, N - 1)] : 0.0f;
    const float bv1 = (mode == 1) ? bias[min(col1, N - 1)] : 0.0f;
#pragma unroll
    for (int i = 0; i < 8; ++i) {
        const int row = rowBase + (half ? (i + 8) : i);
        if (row < M) {
            if (col0 < N) {
                float v = acc0[i];
                if (mode == 1) v = softplus_f(v + bv0);
                C[(size_t)row * ldc + col0] = v;
            }
            if (col1 < N) {
                float v = acc1[i];
                if (mode == 1) v = softplus_f(v + bv1);
                C[(size_t)row * ldc + col1] = v;
            }
        }
    }
}

// ---------------------------------------------------------------------------
// Zero-pad Wdt weights [layers,1152,36] -> [layers,1152,64]
// ---------------------------------------------------------------------------
__global__ void pad_wdt_kernel(const float* __restrict__ src,
                               float* __restrict__ dst, int total)
{
    int idx = blockIdx.x * blockDim.x + threadIdx.x;
    if (idx >= total) return;
    int k = idx % KPAD, rd = idx / KPAD;               // rd = l*1152 + d
    dst[idx] = (k < DTRANK) ? src[(size_t)rd * DTRANK + k] : 0.0f;
}

// ---------------------------------------------------------------------------
// Patch embedding: strided conv3d (2,16,16) + bias + pos/temporal embeds.
// ---------------------------------------------------------------------------
__global__ void patch_embed_kernel(const float* __restrict__ x,
                                   const float* __restrict__ cw,
                                   const float* __restrict__ cb,
                                   const float* __restrict__ pos,   // [197,576]
                                   const float* __restrict__ tpos,  // [4,576]
                                   float* __restrict__ hidden)
{
    int idx = blockIdx.x * blockDim.x + threadIdx.x;
    if (idx >= LSEQ * DMODEL) return;
    int l = idx / DMODEL, c = idx % DMODEL;
    int nt = l / 196, hw = l % 196, hh = hw / 14, ww = hw % 14;
    float acc = cb[c];
    for (int ci = 0; ci < 3; ++ci) {
        for (int kt = 0; kt < 2; ++kt) {
            const float* __restrict__ xs =
                x + (((size_t)ci * 8 + (nt * 2 + kt)) * 224 + hh * 16) * 224 + ww * 16;
            const float* __restrict__ wsp =
                cw + (((size_t)c * 3 + ci) * 2 + kt) * 256;
            for (int py = 0; py < 16; ++py) {
#pragma unroll
                for (int px = 0; px < 16; ++px)
                    acc += xs[py * 224 + px] * wsp[py * 16 + px];
            }
        }
    }
    acc += pos[(size_t)(1 + hw) * DMODEL + c] + tpos[(size_t)nt * DMODEL + c];
    hidden[idx] = acc;
}

__global__ void zero_kernel(float* __restrict__ p, int n)
{
    int i = blockIdx.x * blockDim.x + threadIdx.x;
    if (i < n) p[i] = 0.0f;
}

// ---------------------------------------------------------------------------
// residual += hidden ; hln = LayerNorm(residual)   (one block per row)
// ---------------------------------------------------------------------------
__global__ __launch_bounds__(256)
void add_ln_kernel(const float* __restrict__ hidden,
                   float* __restrict__ residual,
                   float* __restrict__ hln,
                   const float* __restrict__ lw,
                   const float* __restrict__ lb)
{
    const int l = blockIdx.x, tid = threadIdx.x;
    __shared__ float sh0[256], sh1[256];
    float s = 0.0f, s2 = 0.0f;
    for (int c = tid; c < DMODEL; c += 256) {
        float v = hidden[(size_t)l * DMODEL + c] + residual[(size_t)l * DMODEL + c];
        residual[(size_t)l * DMODEL + c] = v;
        s += v; s2 += v * v;
    }
    sh0[tid] = s; sh1[tid] = s2; __syncthreads();
    for (int o = 128; o > 0; o >>= 1) {
        if (tid < o) { sh0[tid] += sh0[tid + o]; sh1[tid] += sh1[tid + o]; }
        __syncthreads();
    }
    const float m   = sh0[0] * (1.0f / DMODEL);
    const float var = sh1[0] * (1.0f / DMODEL) - m * m;
    const float rs  = rsqrtf(var + EPSF);
    for (int c = tid; c < DMODEL; c += 256) {
        float v = residual[(size_t)l * DMODEL + c];
        hln[(size_t)l * DMODEL + c] = (v - m) * rs * lw[c] + lb[c];
    }
}

// ---------------------------------------------------------------------------
// Causal depthwise conv1d (width 4) + bias + SiLU over u = xz[:, :1152].
// rev!=0 reads the sequence reversed; output uc is in *logical* scan order.
// ---------------------------------------------------------------------------
__global__ void conv_silu_kernel(const float* __restrict__ xz,
                                 const float* __restrict__ cw,   // [1152,4]
                                 const float* __restrict__ cb,   // [1152]
                                 float* __restrict__ uc,         // [784,1152]
                                 int rev)
{
    int idx = blockIdx.x * blockDim.x + threadIdx.x;
    if (idx >= LSEQ * DINNER) return;
    int l = idx / DINNER, d = idx % DINNER;
    float acc = cb[d];
#pragma unroll
    for (int j = 0; j < DCONV; ++j) {
        int li = l - (DCONV - 1) + j;
        if (li >= 0) {
            int src = rev ? (LSEQ - 1 - li) : li;
            acc += cw[d * DCONV + j] * xz[(size_t)src * XZDIM + d];
        }
    }
    uc[idx] = silu_f(acc);
}

// ---------------------------------------------------------------------------
// Selective scan, one persistent workgroup per direction.
// 576 threads, 2 channels each, 16 states in registers; B/C via LDS.
// ---------------------------------------------------------------------------
__global__ __launch_bounds__(576)
void scan_kernel(const float* __restrict__ uc,     // [784,1152]
                 const float* __restrict__ delta,  // [784,1152]
                 const float* __restrict__ xdbl,   // [784,XLD] (B:36..51, C:52..67)
                 const float* __restrict__ A_log,  // [1152,16]
                 const float* __restrict__ Dv,     // [1152]
                 float* __restrict__ y)            // [784,1152]
{
    __shared__ float sB[DSTATE], sC[DSTATE];
    const int tid = threadIdx.x;
    const int d0 = tid, d1 = tid + 576;
    float A0[DSTATE], A1[DSTATE], h0[DSTATE], h1[DSTATE];
#pragma unroll
    for (int n = 0; n < DSTATE; ++n) {
        A0[n] = -__expf(A_log[(size_t)d0 * DSTATE + n]);
        A1[n] = -__expf(A_log[(size_t)d1 * DSTATE + n]);
        h0[n] = 0.0f; h1[n] = 0.0f;
    }
    const float Dd0 = Dv[d0], Dd1 = Dv[d1];
    for (int t = 0; t < LSEQ; ++t) {
        if (tid < DSTATE)           sB[tid]          = xdbl[(size_t)t * XLD + DTRANK + tid];
        else if (tid < 2 * DSTATE)  sC[tid - DSTATE] = xdbl[(size_t)t * XLD + DTRANK + tid];
        __syncthreads();
        const float dl0 = delta[(size_t)t * DINNER + d0], u0 = uc[(size_t)t * DINNER + d0];
        const float dl1 = delta[(size_t)t * DINNER + d1], u1 = uc[(size_t)t * DINNER + d1];
        float acc0 = 0.0f, acc1 = 0.0f;
#pragma unroll
        for (int n = 0; n < DSTATE; ++n) {
            const float b = sB[n], c = sC[n];
            h0[n] = __expf(dl0 * A0[n]) * h0[n] + dl0 * b * u0;
            acc0 += h0[n] * c;
            h1[n] = __expf(dl1 * A1[n]) * h1[n] + dl1 * b * u1;
            acc1 += h1[n] * c;
        }
        y[(size_t)t * DINNER + d0] = acc0 + u0 * Dd0;
        y[(size_t)t * DINNER + d1] = acc1 + u1 * Dd1;
        __syncthreads();   // protect sB/sC WAR for next step
    }
}

// ---------------------------------------------------------------------------
// ysum[l,e] = (yf[l,e] + yr[L-1-l,e]) * silu(z[l,e])
// ---------------------------------------------------------------------------
__global__ void combine_kernel(const float* __restrict__ yf,
                               const float* __restrict__ yr,
                               const float* __restrict__ xz,
                               float* __restrict__ ysum)
{
    int idx = blockIdx.x * blockDim.x + threadIdx.x;
    if (idx >= LSEQ * DINNER) return;
    int l = idx / DINNER, e = idx % DINNER;
    float z = xz[(size_t)l * XZDIM + DINNER + e];
    ysum[idx] = (yf[idx] + yr[(size_t)(LSEQ - 1 - l) * DINNER + e]) * silu_f(z);
}

// ---------------------------------------------------------------------------
// out = rms_norm(hidden + residual, w)
// ---------------------------------------------------------------------------
__global__ __launch_bounds__(256)
void final_rms_kernel(const float* __restrict__ hidden,
                      const float* __restrict__ residual,
                      const float* __restrict__ w,
                      float* __restrict__ out)
{
    const int l = blockIdx.x, tid = threadIdx.x;
    __shared__ float sh[256];
    float s2 = 0.0f;
    for (int c = tid; c < DMODEL; c += 256) {
        float v = hidden[(size_t)l * DMODEL + c] + residual[(size_t)l * DMODEL + c];
        s2 += v * v;
    }
    sh[tid] = s2; __syncthreads();
    for (int o = 128; o > 0; o >>= 1) {
        if (tid < o) sh[tid] += sh[tid + o];
        __syncthreads();
    }
    const float rs = rsqrtf(sh[0] * (1.0f / DMODEL) + EPSF);
    for (int c = tid; c < DMODEL; c += 256) {
        float v = hidden[(size_t)l * DMODEL + c] + residual[(size_t)l * DMODEL + c];
        out[(size_t)l * DMODEL + c] = v * rs * w[c];
    }
}

// ===========================================================================
extern "C" void kernel_launch(void* const* d_in, const int* in_sizes, int n_in,
                              void* d_out, int out_size, void* d_ws, size_t ws_size,
                              hipStream_t stream)
{
    const float* x        = (const float*)d_in[0];
    const float* conv_w   = (const float*)d_in[1];
    const float* conv_b   = (const float*)d_in[2];
    const float* pos      = (const float*)d_in[3];
    const float* tpos     = (const float*)d_in[4];
    const float* ln_w     = (const float*)d_in[5];
    const float* ln_b     = (const float*)d_in[6];
    const float* W_in     = (const float*)d_in[7];
    const float* W_out    = (const float*)d_in[8];
    const float* cw_f     = (const float*)d_in[9];
    const float* cb_f     = (const float*)d_in[10];
    const float* Wx_f     = (const float*)d_in[11];
    const float* Wdt_f    = (const float*)d_in[12];
    const float* dtb_f    = (const float*)d_in[13];
    const float* Alog_f   = (const float*)d_in[14];
    const float* D_f      = (const float*)d_in[15];
    const float* cw_r     = (const float*)d_in[16];
    const float* cb_r     = (const float*)d_in[17];
    const float* Wx_r     = (const float*)d_in[18];
    const float* Wdt_r    = (const float*)d_in[19];
    const float* dtb_r    = (const float*)d_in[20];
    const float* Alog_r   = (const float*)d_in[21];
    const float* D_r      = (const float*)d_in[22];
    const float* norm_w   = (const float*)d_in[23];
    float* out = (float*)d_out;

    // ----- workspace layout (fp32 elements, 16-float aligned chunks) -----
    float* ws = (float*)d_ws;
    size_t off = 0;
    auto alloc = [&](size_t n) { float* p = ws + off; off += (n + 63) & ~(size_t)63; return p; };
    float* hidden    = alloc((size_t)LSEQ * DMODEL);
    float* residual  = alloc((size_t)LSEQ * DMODEL);
    float* hln       = alloc((size_t)LSEQ * DMODEL);
    float* xz        = alloc((size_t)LSEQ * XZDIM);
    float* uc_f      = alloc((size_t)LSEQ * DINNER);
    float* uc_r      = alloc((size_t)LSEQ * DINNER);
    float* xdbl_f    = alloc((size_t)LSEQ * XLD);
    float* xdbl_r    = alloc((size_t)LSEQ * XLD);
    float* delta_f   = alloc((size_t)LSEQ * DINNER);
    float* delta_r   = alloc((size_t)LSEQ * DINNER);
    float* yscan_f   = alloc((size_t)LSEQ * DINNER);
    float* yscan_r   = alloc((size_t)LSEQ * DINNER);
    float* ysum      = alloc((size_t)LSEQ * DINNER);
    float* wdtpad_f  = alloc((size_t)DEPTH * DINNER * KPAD);
    float* wdtpad_r  = alloc((size_t)DEPTH * DINNER * KPAD);
    (void)ws_size; (void)in_sizes; (void)n_in; (void)out_size;

    const int nLD = LSEQ * DINNER;                  // 903168
    const dim3 eltBlk(256);
    const dim3 eltGrdLD((nLD + 255) / 256);
    const dim3 eltGrdLM((LSEQ * DMODEL + 255) / 256);

    auto gemm = [&](const float* A, const float* B, float* C, const float* bias,
                    int M, int N, int K, int lda, int ldb, int ldc, int mode) {
        dim3 grid((N + 31) / 32, (M + 63) / 64);
        gemm_wmma_bf16<<<grid, 128, 0, stream>>>(A, B, C, bias, M, N, K, lda, ldb, ldc, mode);
    };

    // ----- one-time-per-call prep: pad Wdt, patch embed, zero residual -----
    {
        int tot = DEPTH * DINNER * KPAD;
        pad_wdt_kernel<<<(tot + 255) / 256, 256, 0, stream>>>(Wdt_f, wdtpad_f, tot);
        pad_wdt_kernel<<<(tot + 255) / 256, 256, 0, stream>>>(Wdt_r, wdtpad_r, tot);
    }
    patch_embed_kernel<<<eltGrdLM, eltBlk, 0, stream>>>(x, conv_w, conv_b, pos, tpos, hidden);
    zero_kernel<<<eltGrdLM, eltBlk, 0, stream>>>(residual, LSEQ * DMODEL);

    // ----- 32 bidirectional Mamba blocks -----
    for (int i = 0; i < DEPTH; ++i) {
        add_ln_kernel<<<LSEQ, 256, 0, stream>>>(hidden, residual, hln,
                                                ln_w + (size_t)i * DMODEL,
                                                ln_b + (size_t)i * DMODEL);
        // xz = hln @ W_in^T     [784,576]x[2304,576]^T
        gemm(hln, W_in + (size_t)i * XZDIM * DMODEL, xz, nullptr,
             LSEQ, XZDIM, DMODEL, DMODEL, DMODEL, XZDIM, 0);

        for (int dir = 0; dir < 2; ++dir) {
            const float* cw   = dir ? cw_r   : cw_f;
            const float* cb   = dir ? cb_r   : cb_f;
            const float* Wx   = dir ? Wx_r   : Wx_f;
            const float* wdtp = dir ? wdtpad_r : wdtpad_f;
            const float* dtb  = dir ? dtb_r  : dtb_f;
            const float* Alog = dir ? Alog_r : Alog_f;
            const float* Dv   = dir ? D_r    : D_f;
            float* uc    = dir ? uc_r    : uc_f;
            float* xdbl  = dir ? xdbl_r  : xdbl_f;
            float* delta = dir ? delta_r : delta_f;
            float* ysc   = dir ? yscan_r : yscan_f;

            conv_silu_kernel<<<eltGrdLD, eltBlk, 0, stream>>>(
                xz, cw + (size_t)i * DINNER * DCONV, cb + (size_t)i * DINNER, uc, dir);
            // xdbl = uc @ Wx^T   [784,1152]x[68,1152]^T   (ldc = 80)
            gemm(uc, Wx + (size_t)i * XDIM * DINNER, xdbl, nullptr,
                 LSEQ, XDIM, DINNER, DINNER, DINNER, XLD, 0);
            // delta = softplus(dt @ WdtPad^T + dtb)  K padded 36->64, zeros kill
            // the B/C columns that dt's padded reads pick up.
            gemm(xdbl, wdtp + (size_t)i * DINNER * KPAD, delta,
                 dtb + (size_t)i * DINNER,
                 LSEQ, DINNER, KPAD, XLD, KPAD, DINNER, 1);
            scan_kernel<<<1, 576, 0, stream>>>(
                uc, delta, xdbl,
                Alog + (size_t)i * DINNER * DSTATE, Dv + (size_t)i * DINNER, ysc);
        }

        combine_kernel<<<eltGrdLD, eltBlk, 0, stream>>>(yscan_f, yscan_r, xz, ysum);
        // hidden = ysum @ W_out^T   [784,1152]x[576,1152]^T
        gemm(ysum, W_out + (size_t)i * DMODEL * DINNER, hidden, nullptr,
             LSEQ, DMODEL, DINNER, DINNER, DINNER, DMODEL, 0);
    }

    final_rms_kernel<<<LSEQ, 256, 0, stream>>>(hidden, residual, norm_w, out);
}
